// LSSN_65944927863180
// MI455X (gfx1250) — matrix operand order repaired and checked by reading, not compiled, not run on hardware
//
#include <hip/hip_runtime.h>

typedef __attribute__((ext_vector_type(16))) _Float16 v16h;
typedef __attribute__((ext_vector_type(8)))  _Float16 v8h;
typedef __attribute__((ext_vector_type(8)))  float    v8f;

#define N_IN   512
#define N_REC  2048
#define N_OUT  10
#define KTOT   2560          // N_REC + N_IN
#define BATCH  32
#define TSTEPS 500

__device__ __constant__ float C_ALPHA = 0.9512294245007140f;  // exp(-1/20)
__device__ __constant__ float C_RHO   = 0.9995001249791693f;  // exp(-1/2000)
__device__ __constant__ float C_KAPPA = 0.3678794411714423f;  // exp(-1)
#define C_THR   0.6f
#define C_BETA  1.7f
#define C_REFR  5.0f

// ---------------- one-time prep kernels ----------------

// WT[n][k] (f16) = k<2048 ? w_rec[k][n] : w_in[k-2048][n]
__global__ void LSSN_prep_weights(const float* __restrict__ w_rec,
                                  const float* __restrict__ w_in,
                                  _Float16* __restrict__ WT) {
  int id = blockIdx.x * 256 + threadIdx.x;
  if (id >= N_REC * KTOT) return;
  int n = id / KTOT;
  int k = id - n * KTOT;
  float v = (k < N_REC) ? w_rec[(size_t)k * N_REC + n]
                        : w_in[(size_t)(k - N_REC) * N_REC + n];
  WT[id] = (_Float16)v;
}

__global__ void LSSN_prep_x(const float* __restrict__ x, _Float16* __restrict__ xh, int total) {
  int id = blockIdx.x * 256 + threadIdx.x;
  if (id < total) xh[id] = (_Float16)x[id];
}

__global__ void LSSN_prep_state(float* __restrict__ sv, float* __restrict__ sa,
                                float* __restrict__ sr, _Float16* __restrict__ zh0) {
  int id = blockIdx.x * 256 + threadIdx.x;
  if (id < BATCH * N_REC) { sv[id] = 0.f; sa[id] = 0.f; sr[id] = 0.f; zh0[id] = (_Float16)0.f; }
}

// ---------------- per-timestep fused GEMM + neuron update ----------------
// 128 waves (16 blocks x 8 waves). Wave w owns output columns [16w,16w+16) for
// both batch row-tiles (rows 0..15 and 16..31): two f32 16x16 accumulators.

__device__ __forceinline__ void LSSN_ld_chunk(const _Float16* __restrict__ b,
                                              const _Float16* __restrict__ a0,
                                              const _Float16* __restrict__ a1,
                                              v16h& bm, v16h& am0, v16h& am1) {
  v8h blo = *(const v8h*)(b);
  v8h bhi = *(const v8h*)(b + 16);
  v8h a0lo = *(const v8h*)(a0);
  v8h a0hi = *(const v8h*)(a0 + 16);
  v8h a1lo = *(const v8h*)(a1);
  v8h a1hi = *(const v8h*)(a1 + 16);
#pragma unroll
  for (int i = 0; i < 8; ++i) {
    bm[i] = blo[i]; bm[i + 8] = bhi[i];
    am0[i] = a0lo[i]; am0[i + 8] = a0hi[i];
    am1[i] = a1lo[i]; am1[i + 8] = a1hi[i];
  }
}

__global__ __launch_bounds__(256) void LSSN_alif_step(
    const _Float16* __restrict__ WT,     // [N_REC][KTOT]
    const _Float16* __restrict__ xh,     // [B][T][N_IN]
    const _Float16* __restrict__ zh_in,  // [B][N_REC]  (z_{t-1})
    _Float16* __restrict__ zh_out,       // [B][N_REC]  (z_t)
    float* __restrict__ st_v, float* __restrict__ st_a, float* __restrict__ st_r,
    float* __restrict__ z_hist,          // [B][T][N_REC]
    int t)
{
  const int tid  = threadIdx.x;
  const int lane = tid & 31;
  const int wave = (blockIdx.x * 256 + tid) >> 5;   // 0..127
  const int n0   = wave * 16;
  const int lm   = lane & 15;
  const int hi   = lane >> 4;        // half-wave select
  const int klo  = hi * 8;           // A: this lane's K sub-offset
  const int kbB  = hi * 8;           // B: this lane's K sub-offset

  v8f c0 = {};   // rows 0..15
  v8f c1 = {};   // rows 16..31

  // loop-invariant base pointers (no control flow in the K loops)
  const _Float16* __restrict__ bp  = WT + (size_t)(n0 + lm) * KTOT + kbB;
  const _Float16* __restrict__ z0p = zh_in + (size_t)lm * N_REC + klo;
  const _Float16* __restrict__ z1p = zh_in + (size_t)(lm + 16) * N_REC + klo;
  const _Float16* __restrict__ x0p = xh + ((size_t)lm * TSTEPS + t) * N_IN + klo;
  const _Float16* __restrict__ x1p = xh + ((size_t)(lm + 16) * TSTEPS + t) * N_IN + klo;

  v16h bm, am0, am1;

  // software pipeline: loads for chunk i+1 are issued before the WMMAs of chunk i
  LSSN_ld_chunk(bp, z0p, z1p, bm, am0, am1);

#pragma unroll 2
  for (int k0 = 32; k0 < N_REC; k0 += 32) {
    v16h bn, an0, an1;
    LSSN_ld_chunk(bp + k0, z0p + k0, z1p + k0, bn, an0, an1);
    c0 = __builtin_amdgcn_wmma_f32_16x16x32_f16(false, am0, false, bm, (short)0, c0, false, false);
    c1 = __builtin_amdgcn_wmma_f32_16x16x32_f16(false, am1, false, bm, (short)0, c1, false, false);
    bm = bn; am0 = an0; am1 = an1;
  }
  {  // bridge: consume last z chunk, prefetch first x chunk
    v16h bn, an0, an1;
    LSSN_ld_chunk(bp + N_REC, x0p, x1p, bn, an0, an1);
    c0 = __builtin_amdgcn_wmma_f32_16x16x32_f16(false, am0, false, bm, (short)0, c0, false, false);
    c1 = __builtin_amdgcn_wmma_f32_16x16x32_f16(false, am1, false, bm, (short)0, c1, false, false);
    bm = bn; am0 = an0; am1 = an1;
  }
#pragma unroll 2
  for (int kx = 32; kx < N_IN; kx += 32) {
    v16h bn, an0, an1;
    LSSN_ld_chunk(bp + N_REC + kx, x0p + kx, x1p + kx, bn, an0, an1);
    c0 = __builtin_amdgcn_wmma_f32_16x16x32_f16(false, am0, false, bm, (short)0, c0, false, false);
    c1 = __builtin_amdgcn_wmma_f32_16x16x32_f16(false, am1, false, bm, (short)0, c1, false, false);
    bm = bn; am0 = an0; am1 = an1;
  }
  // drain
  c0 = __builtin_amdgcn_wmma_f32_16x16x32_f16(false, am0, false, bm, (short)0, c0, false, false);
  c1 = __builtin_amdgcn_wmma_f32_16x16x32_f16(false, am1, false, bm, (short)0, c1, false, false);

  // ---- fused ALIF neuron dynamics on the 2x(16x16) tile ----
  const int   n      = n0 + lm;
  const float beta_n = (n >= 1024) ? C_BETA : 0.0f;
  const float alpha  = C_ALPHA, rho = C_RHO;

#pragma unroll
  for (int rb = 0; rb < 2; ++rb) {
    v8f c = rb ? c1 : c0;
#pragma unroll
    for (int i = 0; i < 8; ++i) {
      // f32 C/D layout: VGPR i -> M = i + 8*hi (+16 for second row tile), N = lane&15
      int   m   = rb * 16 + hi * 8 + i;
      int   idx = m * N_REC + n;
      float pre = c[i];
      float zp  = (float)zh_in[idx];
      float r   = st_r[idx];
      float nr  = (zp > 0.f) ? C_REFR : fmaxf(0.f, r - 1.f);
      float vv  = st_v[idx];
      float nv  = vv * (alpha - C_THR) + (1.f - alpha) * pre;   // a*v+(1-a)*pre - v*thr
      float aa  = st_a[idx];
      float na  = rho * aa + (1.f - rho) * zp;
      float athr = C_THR + na * beta_n;
      float vs   = (nv - athr) / C_THR;
      float z    = (vs > 0.f && nr == 0.f) ? 1.f : 0.f;
      st_v[idx] = nv;
      st_a[idx] = na;
      st_r[idx] = nr;
      zh_out[idx] = (_Float16)z;
      // stream spike history to HBM without polluting L2 (weights live there)
      __builtin_nontemporal_store(z, &z_hist[((size_t)m * TSTEPS + t) * N_REC + n]);
    }
  }
}

// ---------------- readout: y_t = k*y_{t-1} + (1-k) * z_t @ w_out ----------------
// one wave per (b,o) pair; lanes split the N_REC reduction.
__global__ __launch_bounds__(256) void LSSN_readout(
    const float* __restrict__ z_hist, const float* __restrict__ w_out,
    float* __restrict__ y_out)
{
  int wave = (blockIdx.x * 256 + threadIdx.x) >> 5;   // 0..319
  int lane = threadIdx.x & 31;
  if (wave >= BATCH * N_OUT) return;
  int b = wave / N_OUT;
  int o = wave - b * N_OUT;

  float wc[64];
#pragma unroll
  for (int j = 0; j < 64; ++j) wc[j] = w_out[(size_t)(lane + 32 * j) * N_OUT + o];

  const float kap = C_KAPPA;
  float y = 0.f;
  const float* zb = z_hist + (size_t)b * TSTEPS * N_REC;
  for (int t = 0; t < TSTEPS; ++t) {
    const float* zr = zb + (size_t)t * N_REC;
    float p = 0.f;
#pragma unroll
    for (int j = 0; j < 64; ++j)
      p += __builtin_nontemporal_load(&zr[lane + 32 * j]) * wc[j];
#pragma unroll
    for (int off = 16; off; off >>= 1) p += __shfl_xor(p, off, 32);
    y = kap * y + (1.f - kap) * p;
    if (lane == 0) y_out[((size_t)b * TSTEPS + t) * N_OUT + o] = y;
  }
}

// ---------------- launch ----------------
extern "C" void kernel_launch(void* const* d_in, const int* in_sizes, int n_in,
                              void* d_out, int out_size, void* d_ws, size_t ws_size,
                              hipStream_t stream) {
  const float* x     = (const float*)d_in[0];   // (32,500,512)
  const float* w_in  = (const float*)d_in[1];   // (512,2048)
  const float* w_rec = (const float*)d_in[2];   // (2048,2048)
  const float* w_out = (const float*)d_in[3];   // (2048,10)

  float* y_out = (float*)d_out;                                   // (32,500,10)
  float* z_out = (float*)d_out + (size_t)BATCH * TSTEPS * N_OUT;  // (32,500,2048)

  // workspace carve-up (256B aligned)
  char* ws = (char*)d_ws;
  size_t off = 0;
  auto carve = [&](size_t bytes) -> void* {
    void* p = ws + off;
    off = (off + bytes + 255) & ~(size_t)255;
    return p;
  };
  _Float16* WT   = (_Float16*)carve((size_t)N_REC * KTOT * sizeof(_Float16));          // 10.5 MB
  _Float16* xh   = (_Float16*)carve((size_t)BATCH * TSTEPS * N_IN * sizeof(_Float16)); // 16.4 MB
  _Float16* zh   = (_Float16*)carve((size_t)2 * BATCH * N_REC * sizeof(_Float16));     // double buffer
  float*    st_v = (float*)carve((size_t)BATCH * N_REC * sizeof(float));
  float*    st_a = (float*)carve((size_t)BATCH * N_REC * sizeof(float));
  float*    st_r = (float*)carve((size_t)BATCH * N_REC * sizeof(float));
  if (off > ws_size) return;  // workspace too small; nothing safe to do

  const int totW = N_REC * KTOT;
  const int totX = BATCH * TSTEPS * N_IN;
  LSSN_prep_weights<<<(totW + 255) / 256, 256, 0, stream>>>(w_rec, w_in, WT);
  LSSN_prep_x<<<(totX + 255) / 256, 256, 0, stream>>>(x, xh, totX);
  LSSN_prep_state<<<(BATCH * N_REC + 255) / 256, 256, 0, stream>>>(st_v, st_a, st_r, zh);

  for (int t = 0; t < TSTEPS; ++t) {
    const _Float16* zin = zh + (size_t)(t & 1) * (BATCH * N_REC);
    _Float16* zout      = zh + (size_t)((t + 1) & 1) * (BATCH * N_REC);
    LSSN_alif_step<<<16, 256, 0, stream>>>(WT, xh, zin, zout, st_v, st_a, st_r, z_out, t);
  }

  LSSN_readout<<<40, 256, 0, stream>>>(z_out, w_out, y_out);
}
